// _SequenceLSTM_26877905338383
// MI455X (gfx1250) — matrix operand order repaired and checked by reading, not compiled, not run on hardware
//
#include <hip/hip_runtime.h>

#define LSEQ 512
#define NB   2048
#define HID  20

typedef __attribute__((ext_vector_type(16))) _Float16 v16h;
typedef __attribute__((ext_vector_type(8)))  float    v8f;

union V8 { v8f v; float f[8]; };

// ---- A-fragment tile offsets (each tile = 512 halves = 16x32 f16) ----
#define OFF_W0   0
#define OFF_W1   2
#define OFF_WIH1 4
#define OFF_WHH1 12
#define OFF_WIH2 20
#define OFF_WHH2 28
#define OFF_W2   36
#define OFF_W3   38
#define OFF_W4   40
#define OFF_HEAD 42
#define NTILES   43

// ---- f32 shared scalar offsets ----
#define B0o   0
#define B1o   32
#define BL1o  64
#define BL2o  192
#define B2o   320
#define B3o   352
#define B4o   384
#define BHo   416
#define LN1o  432
#define LNAo  464
#define LNBWo 496
#define LNBBo 528
#define LNCo  560
#define LNDo  592
#define LNEo  624
#define MSo   656
#define NSF   660

struct KP {
  const float *x;
  const float *mean, *std_;
  const float *w0, *b0, *ln1;
  const float *w1, *b1, *lnA;
  const float *wih1, *whh1, *bih1, *bhh1;
  const float *lnBw, *lnBb;
  const float *wih2, *whh2, *bih2, *bhh2;
  const float *lnC, *w2, *b2;
  const float *lnD, *w3, *b3;
  const float *lnE, *w4, *b4;
  const float *wfcw, *wfcb, *sfcw, *sfcb, *dfcw, *dfcb;
  float *out;
};

// ---- fast transcendentals: raw gfx1250 TRANS ops ----
#define LOG2E 1.44269504088896f
#define LN2   0.69314718055995f
__device__ __forceinline__ float fexp (float x){ return __builtin_amdgcn_exp2f(x * LOG2E); }
__device__ __forceinline__ float frcp (float x){ return __builtin_amdgcn_rcpf(x); }
__device__ __forceinline__ float frsq (float x){ return __builtin_amdgcn_rsqf(x); }
__device__ __forceinline__ float flog (float x){ return __builtin_amdgcn_logf(x) * LN2; }

// gfx1250 has V_TANH_F32 (TRANS32). Use it if clang exposes the builtin;
// otherwise fall back to exp2/rcp form.
#if __has_builtin(__builtin_amdgcn_tanhf)
__device__ __forceinline__ float ftanh(float x){ return __builtin_amdgcn_tanhf(x); }
#else
__device__ __forceinline__ float ftanh(float x){ return 1.0f - 2.0f*frcp(1.0f + fexp(2.0f*x)); }
#endif
// sigmoid(x) = 0.5 + 0.5*tanh(x/2): mul + tanh + fma, no reciprocal
__device__ __forceinline__ float sigm (float x){ return fmaf(0.5f, ftanh(0.5f*x), 0.5f); }
__device__ __forceinline__ float silu (float x){ return x * sigm(x); }

// ---- stage a weight matrix as WMMA A-fragments (M=out unit, K=in unit padded to 32)
// A layout (16-bit, 16x32): lane<16: M=lane, halves 0..7 = K 0..7, halves 8..15 = K 16..23
//                           lane>=16: M=lane-16, K 8..15 / 24..31.
__device__ void stageA(_Float16* dst, const float* W, int rows, int cols, int mtiles,
                       bool lstmReorder, int tid, int nt){
  int tot = mtiles*512;
  for (int e=tid; e<tot; e+=nt){
    int lane=(e>>4)&31, h=e&15;
    int M = ((e>>9)<<4) + (lane&15);
    int K = (h<8 ? h : h+8) + ((lane>=16)?8:0);
    int src;
    if (lstmReorder){ int r=M&31; src = (r<HID)? (M>>5)*HID + r : -1; }   // gate blocks padded 20->32
    else            { src = (M<rows)? M : -1; }
    dst[e] = (_Float16)((src>=0 && K<cols)? W[src*cols+K] : 0.0f);
  }
}

__device__ void stageHead(_Float16* dst, const float* wfc, const float* sfc, const float* dfc,
                          int tid, int nt){
  for (int e=tid; e<512; e+=nt){
    int lane=(e>>4)&31, h=e&15;
    int M = lane&15;
    int K = (h<8 ? h : h+8) + ((lane>=16)?8:0);
    float v = 0.0f;
    if (M<9 && K<HID){
      const float* W = (M<3)? wfc : (M<6 ? sfc : dfc);
      v = W[(M%3)*HID + K];
    }
    dst[e] = (_Float16)v;
  }
}

__device__ void stageBiasPlain(float* dst, const float* b, int nreal, int npad, int tid, int nt){
  for (int i=tid;i<npad;i+=nt) dst[i] = (i<nreal)? b[i] : 0.0f;
}
__device__ void stageBiasLstm(float* dst, const float* b1, const float* b2, int tid, int nt){
  for (int p=tid;p<128;p+=nt){
    int r=p&31, g=p>>5;
    dst[p] = (r<HID)? (b1[g*HID+r]+b2[g*HID+r]) : 0.0f;
  }
}

// ---- one 16x16x32 WMMA with A from LDS fragment store ----
__device__ __forceinline__ v8f wmma_t(const _Float16* frag, int tile, int lane, v16h b, v8f c){
  const v16h a = *(const v16h*)(frag + (tile<<9) + (lane<<4));
  return __builtin_amdgcn_wmma_f32_16x16x32_f16(false, a, false, b, (short)0, c, false, false);
}

// ---- build B fragment (K=unit, N=sample) from lane-local slots.
// Slots: lane<16 owns units {0..7,16..19}; lane>=16 owns units {8..15} (slots 8..11 == 0).
// B layout: lane<16 halves = K 0..7,16..23; lane>=16 halves = K 8..15,24..31.
__device__ __forceinline__ v16h packB_swap(const float* v, bool hi){
  float send[8], recv[8];
  #pragma unroll
  for (int k=0;k<8;k++) send[k] = hi ? v[k] : (k<4 ? v[8+k] : 0.0f);
  #pragma unroll
  for (int k=0;k<8;k++) recv[k] = __shfl_xor(send[k], 16, 32);
  v16h b;
  #pragma unroll
  for (int k=0;k<8;k++) b[k]   = (_Float16)(v[k]);
  #pragma unroll
  for (int k=0;k<8;k++) b[8+k] = (_Float16)(hi ? 0.0f : (k<4 ? v[8+k] : 0.0f));
  return b;
}

// ---- LayerNorm over the 20 real units (pads are 0); one half-wave shuffle pair ----
__device__ __forceinline__ void lnrm(float* v, const float* w, const float* b, bool hi){
  float s=0.f, s2=0.f;
  #pragma unroll
  for (int j=0;j<12;j++){ s += v[j]; s2 += v[j]*v[j]; }
  s  += __shfl_xor(s, 16, 32);
  s2 += __shfl_xor(s2, 16, 32);
  float m  = s * (1.0f/HID);
  float vr = s2 * (1.0f/HID) - m*m;
  float rs = frsq(vr + 1e-5f);
  #pragma unroll
  for (int j=0;j<12;j++){
    int u = j + (j<8?0:8) + (hi?8:0);          // padded arrays are 0 past unit 19
    v[j] = (v[j]-m)*rs*w[u] + (b ? b[u] : 0.0f);
  }
}

// ---- 2-tile (32 padded out units) dense layer: D slots match state slot layout ----
__device__ __forceinline__ void layer2(const _Float16* sA, const float* sF, int TO, int BO,
                                       int lane, bool hi, v16h b, float* out){
  int base0 = BO + (hi?8:0), base1 = BO + 16 + (hi?8:0);
  v8f c0, c1;
  #pragma unroll
  for (int j=0;j<8;j++){ c0[j]=sF[base0+j]; c1[j]=sF[base1+j]; }
  c0 = wmma_t(sA, TO+0, lane, b, c0);
  c1 = wmma_t(sA, TO+1, lane, b, c1);
  V8 u0,u1; u0.v=c0; u1.v=c1;
  #pragma unroll
  for (int k=0;k<8;k++) out[k]   = u0.f[k];
  #pragma unroll
  for (int k=0;k<4;k++) out[8+k] = u1.f[k];
}

// ---- one LSTM timestep: 16 WMMAs, lane-local gate math ----
__device__ __forceinline__ void lstm_step(const _Float16* sA, const float* sF,
    int TOih, int TOhh, int BO, int lane, bool hi,
    const float* xv, float* hs, float* cs){
  v16h xb = packB_swap(xv, hi);
  v16h hb = packB_swap(hs, hi);
  V8 acc[8];
  #pragma unroll
  for (int t=0;t<8;t++){
    int base = BO + t*16 + (hi?8:0);
    v8f c;
    #pragma unroll
    for (int j=0;j<8;j++) c[j]=sF[base+j];
    c = wmma_t(sA, TOih+t, lane, xb, c);
    c = wmma_t(sA, TOhh+t, lane, hb, c);
    acc[t].v = c;
  }
  #pragma unroll
  for (int j=0;j<12;j++){
    int tk = (j<8)?0:1, k = (j<8)? j : j-8;
    float gi = acc[0+tk].f[k];
    float gf = acc[2+tk].f[k];
    float gg = acc[4+tk].f[k];
    float go = acc[6+tk].f[k];
    float c  = sigm(gf)*cs[j] + sigm(gi)*ftanh(gg);
    cs[j] = c;
    hs[j] = sigm(go)*ftanh(c);
  }
}

__launch_bounds__(64)
__global__ void fused_seq_lstm(KP p){
  const int tid  = threadIdx.x;
  const int lane = tid & 31;
  const bool hi  = lane >= 16;
  const int wave = blockIdx.x*2 + (tid>>5);
  const int n    = wave*16 + (lane & 15);

  __shared__ __align__(32) _Float16 sA[NTILES*512];
  __shared__ float sF[NSF];

  // ---- stage all weights/biases once ----
  stageA(sA + OFF_W0*512,   p.w0,   20, 6,  2, false, tid, 64);
  stageA(sA + OFF_W1*512,   p.w1,   20, 20, 2, false, tid, 64);
  stageA(sA + OFF_WIH1*512, p.wih1, 80, 20, 8, true,  tid, 64);
  stageA(sA + OFF_WHH1*512, p.whh1, 80, 20, 8, true,  tid, 64);
  stageA(sA + OFF_WIH2*512, p.wih2, 80, 20, 8, true,  tid, 64);
  stageA(sA + OFF_WHH2*512, p.whh2, 80, 20, 8, true,  tid, 64);
  stageA(sA + OFF_W2*512,   p.w2,   20, 20, 2, false, tid, 64);
  stageA(sA + OFF_W3*512,   p.w3,   20, 20, 2, false, tid, 64);
  stageA(sA + OFF_W4*512,   p.w4,   20, 20, 2, false, tid, 64);
  stageHead(sA + OFF_HEAD*512, p.wfcw, p.sfcw, p.dfcw, tid, 64);

  stageBiasPlain(sF+B0o, p.b0, 20, 32, tid, 64);
  stageBiasPlain(sF+B1o, p.b1, 20, 32, tid, 64);
  stageBiasLstm (sF+BL1o, p.bih1, p.bhh1, tid, 64);
  stageBiasLstm (sF+BL2o, p.bih2, p.bhh2, tid, 64);
  stageBiasPlain(sF+B2o, p.b2, 20, 32, tid, 64);
  stageBiasPlain(sF+B3o, p.b3, 20, 32, tid, 64);
  stageBiasPlain(sF+B4o, p.b4, 20, 32, tid, 64);
  for (int i=tid;i<16;i+=64)
    sF[BHo+i] = (i<3)? p.wfcb[i] : (i<6)? p.sfcb[i-3] : (i<9)? p.dfcb[i-6] : 0.0f;
  stageBiasPlain(sF+LN1o,  p.ln1,  20, 32, tid, 64);
  stageBiasPlain(sF+LNAo,  p.lnA,  20, 32, tid, 64);
  stageBiasPlain(sF+LNBWo, p.lnBw, 20, 32, tid, 64);
  stageBiasPlain(sF+LNBBo, p.lnBb, 20, 32, tid, 64);
  stageBiasPlain(sF+LNCo,  p.lnC,  20, 32, tid, 64);
  stageBiasPlain(sF+LNDo,  p.lnD,  20, 32, tid, 64);
  stageBiasPlain(sF+LNEo,  p.lnE,  20, 32, tid, 64);
  if (tid<4) sF[MSo+tid] = (tid<2)? p.mean[tid] : p.std_[tid-2];
  __syncthreads();

  const float ms0=sF[MSo+0], ms1=sF[MSo+1];
  const float isd0=frcp(sF[MSo+2]), isd1=frcp(sF[MSo+3]);   // reciprocal std, hoisted

  float h1[12], c1[12], h2[12], c2[12];
  #pragma unroll
  for (int j=0;j<12;j++){ h1[j]=0.f; c1[j]=0.f; h2[j]=0.f; c2[j]=0.f; }

  float* outW = p.out;
  float* outS = p.out + (size_t)LSEQ*NB*3;
  float* outD = p.out + (size_t)LSEQ*NB*3*2;

  for (int t=0; t<LSEQ; ++t){
    // -------- feature construction (lo half-wave loads its sample) --------
    float feat[6];
    #pragma unroll
    for (int k=0;k<6;k++) feat[k]=0.0f;
    if (!hi){
      const float* xp = p.x + ((size_t)t*NB + n)*3;
      float delay=xp[0], dur=xp[1], rat=xp[2];
      feat[0] = (flog(1e-7f + delay) - ms0)*isd0;
      float dcl = fminf(fmaxf(dur, 100.0f), 60000.0f);
      feat[1] = (flog(dcl) - ms1)*isd1;
      int idx = (int)fmaxf(rat, 1.0f) - 1;
      #pragma unroll
      for (int k=0;k<4;k++) feat[2+k] = (idx==k)? 1.0f : 0.0f;
    }
    v16h xb;
    #pragma unroll
    for (int k=0;k<16;k++) xb[k] = (_Float16)((!hi && k<6)? feat[k] : 0.0f);

    // -------- MLP front: silu(w0) -> ln1 -> silu(w1) --------
    float hv[12];
    layer2(sA,sF,OFF_W0,B0o,lane,hi,xb,hv);
    #pragma unroll
    for (int j=0;j<12;j++) hv[j]=silu(hv[j]);
    lnrm(hv, sF+LN1o, nullptr, hi);
    float h[12];
    layer2(sA,sF,OFF_W1,B1o,lane,hi,packB_swap(hv,hi),h);
    #pragma unroll
    for (int j=0;j<12;j++) h[j]=silu(h[j]);

    // -------- LSTM 1 (residual) --------
    float xl[12];
    #pragma unroll
    for (int j=0;j<12;j++) xl[j]=h[j];
    lnrm(xl, sF+LNAo, nullptr, hi);
    lstm_step(sA,sF,OFF_WIH1,OFF_WHH1,BL1o,lane,hi,xl,h1,c1);
    #pragma unroll
    for (int j=0;j<12;j++) h[j]+=h1[j];

    // -------- LSTM 2 (residual) --------
    float xl2[12];
    #pragma unroll
    for (int j=0;j<12;j++) xl2[j]=h[j];
    lnrm(xl2, sF+LNBWo, sF+LNBBo, hi);
    lstm_step(sA,sF,OFF_WIH2,OFF_WHH2,BL2o,lane,hi,xl2,h2,c2);
    #pragma unroll
    for (int j=0;j<12;j++) h[j]+=h2[j];

    // -------- post MLP: t = silu(w3(lnD(silu(w2(lnC(h)))))) ; h += t --------
    float tb[12];
    #pragma unroll
    for (int j=0;j<12;j++) tb[j]=h[j];
    lnrm(tb, sF+LNCo, nullptr, hi);
    float t2[12];
    layer2(sA,sF,OFF_W2,B2o,lane,hi,packB_swap(tb,hi),t2);
    #pragma unroll
    for (int j=0;j<12;j++) t2[j]=silu(t2[j]);
    lnrm(t2, sF+LNDo, nullptr, hi);
    float t3[12];
    layer2(sA,sF,OFF_W3,B3o,lane,hi,packB_swap(t2,hi),t3);
    #pragma unroll
    for (int j=0;j<12;j++) h[j]+=silu(t3[j]);

    // -------- final: hF = silu(w4(lnE(h))) --------
    float e[12];
    #pragma unroll
    for (int j=0;j<12;j++) e[j]=h[j];
    lnrm(e, sF+LNEo, nullptr, hi);
    float hf[12];
    layer2(sA,sF,OFF_W4,B4o,lane,hi,packB_swap(e,hi),hf);
    #pragma unroll
    for (int j=0;j<12;j++) hf[j]=silu(hf[j]);

    // -------- fused 9-row head tile: rows {wfc0..2, sfc0..2, dfc0..2} --------
    v8f hc;
    {
      int base = BHo + (hi?8:0);
      #pragma unroll
      for (int j=0;j<8;j++) hc[j]=sF[base+j];
    }
    hc = wmma_t(sA, OFF_HEAD, lane, packB_swap(hf,hi), hc);
    V8 hu; hu.v = hc;

    size_t ob = ((size_t)t*NB + n)*3;
    if (!hi){
      float w0l=hu.f[0], w1l=hu.f[1], w2l=hu.f[2];
      float mx = fmaxf(w0l, fmaxf(w1l, w2l));
      float e0=fexp(w0l-mx), e1=fexp(w1l-mx), e2=fexp(w2l-mx);
      float inv = frcp(e0+e1+e2);
      outW[ob+0]=e0*inv; outW[ob+1]=e1*inv; outW[ob+2]=e2*inv;
      outS[ob+0]=fexp(fminf(fmaxf(hu.f[3],-25.f),25.f));
      outS[ob+1]=fexp(fminf(fmaxf(hu.f[4],-25.f),25.f));
      outS[ob+2]=fexp(fminf(fmaxf(hu.f[5],-25.f),25.f));
      outD[ob+0]=fexp(fminf(fmaxf(hu.f[6],-25.f),25.f));
      outD[ob+1]=fexp(fminf(fmaxf(hu.f[7],-25.f),25.f));
    } else {
      outD[ob+2]=fexp(fminf(fmaxf(hu.f[0],-25.f),25.f));
    }
  }
}

extern "C" void kernel_launch(void* const* d_in, const int* in_sizes, int n_in,
                              void* d_out, int out_size, void* d_ws, size_t ws_size,
                              hipStream_t stream){
  (void)in_sizes; (void)n_in; (void)out_size; (void)d_ws; (void)ws_size;
  KP p;
  p.x    = (const float*)d_in[0];
  p.mean = (const float*)d_in[1];  p.std_ = (const float*)d_in[2];
  p.w0   = (const float*)d_in[3];  p.b0   = (const float*)d_in[4];  p.ln1 = (const float*)d_in[5];
  p.w1   = (const float*)d_in[6];  p.b1   = (const float*)d_in[7];  p.lnA = (const float*)d_in[8];
  p.wih1 = (const float*)d_in[9];  p.whh1 = (const float*)d_in[10];
  p.bih1 = (const float*)d_in[11]; p.bhh1 = (const float*)d_in[12];
  p.lnBw = (const float*)d_in[13]; p.lnBb = (const float*)d_in[14];
  p.wih2 = (const float*)d_in[15]; p.whh2 = (const float*)d_in[16];
  p.bih2 = (const float*)d_in[17]; p.bhh2 = (const float*)d_in[18];
  p.lnC  = (const float*)d_in[19]; p.w2   = (const float*)d_in[20]; p.b2 = (const float*)d_in[21];
  p.lnD  = (const float*)d_in[22]; p.w3   = (const float*)d_in[23]; p.b3 = (const float*)d_in[24];
  p.lnE  = (const float*)d_in[25]; p.w4   = (const float*)d_in[26]; p.b4 = (const float*)d_in[27];
  p.wfcw = (const float*)d_in[28]; p.wfcb = (const float*)d_in[29];
  p.sfcw = (const float*)d_in[30]; p.sfcb = (const float*)d_in[31];
  p.dfcw = (const float*)d_in[32]; p.dfcb = (const float*)d_in[33];
  p.out  = (float*)d_out;

  // 2048 samples / 16 per wave = 128 waves; 2 waves per block -> 64 blocks
  fused_seq_lstm<<<64, 64, 0, stream>>>(p);
}
